// RNNTaggerModel_37701222924573
// MI455X (gfx1250) — matrix operand order repaired and checked by reading, not compiled, not run on hardware
//
#include <hip/hip_runtime.h>
#include <hip/hip_bf16.h>

typedef __attribute__((ext_vector_type(16))) _Float16 v16h;
typedef __attribute__((ext_vector_type(8)))  float    v8f;
typedef __attribute__((ext_vector_type(4)))  float    v4f;

// Problem constants (fixed by the reference harness)
constexpr int Bsz = 128;
constexpr int Tsz = 512;
constexpr int Lsz = 256;
constexpr float NEG_INF = -1e4f;

// ---------------------------------------------------------------------------
// Forward algorithm: log-semiring matvec recast as linear GEMM on WMMA.
//   alpha'[b,i] = m_b + log( sum_j exp(alpha[b,j]-m_b) * exp(trans[i,j]) ) + u
// One block = 16 batch rows, 8 waves, full T loop. expT resident in LDS (f16).
// kb-outer loop: A fragment loaded once per k-step, shared by 2 column tiles.
// ---------------------------------------------------------------------------
__global__ __launch_bounds__(256)
void crf_forward_wmma(const float* __restrict__ unary,   // [B,T,L]
                      const float* __restrict__ trans,   // [L,L]  (to, from)
                      const int*   __restrict__ lengths, // [B]
                      const int*   __restrict__ sidx_p,
                      const int*   __restrict__ eidx_p,
                      float*       __restrict__ fwd_out) // [B]
{
    __shared__ _Float16 expT[Lsz * Lsz];   // [i][j] = exp(trans[i][j])  (128 KB)
    __shared__ float    alpha[16][Lsz];    // 16 KB
    __shared__ _Float16 pbuf[16][Lsz];     // 8 KB, p = exp(alpha - m)
    __shared__ float    red[16][16];
    __shared__ float    mrow[16];
    __shared__ int      slen[16];

    const int tid  = threadIdx.x;        // 0..255
    const int lane = tid & 31;
    const int wave = tid >> 5;           // 0..7
    const int b0   = blockIdx.x * 16;
    const int start_idx = sidx_p[0];
    const int end_idx   = eidx_p[0];

    // Pin exp(transitions) in LDS as f16 (B-matrix source, contiguous in K=j).
    for (int idx = tid; idx < Lsz * Lsz; idx += 256)
        expT[idx] = (_Float16)__expf(trans[idx]);

    // alpha init: NEG_INF except start tag.
    for (int idx = tid; idx < 16 * Lsz; idx += 256) {
        int r = idx >> 8, c = idx & 255;
        alpha[r][c] = (c == start_idx) ? 0.0f : NEG_INF;
    }
    if (tid < 16) slen[tid] = lengths[b0 + tid];
    __syncthreads();

    // Per-lane constant fragment coordinates (ISA 7.12.2 layouts).
    const int arow  = lane & 15;                 // A: M = lane % 16
    const int koff  = (lane < 16) ? 0 : 8;       // A: K half-offset
    const int bko   = (lane < 16) ? 0 : 8;       // B: K half-offset
    const int ncol0 = (wave * 2) * 16 + (lane & 15);
    const int ncol1 = ncol0 + 16;
    const int rbase = (lane < 16) ? 0 : 8;       // C/D: M = r + rbase

    for (int t = 0; t < Tsz; ++t) {
        // ---- row max (16 threads per batch row) ----
        {
            int r = tid >> 4, p = tid & 15;
            float m = -3.4e38f;
            for (int j = p; j < Lsz; j += 16) m = fmaxf(m, alpha[r][j]);
            red[r][p] = m;
        }
        __syncthreads();
        if (tid < 16) {
            float m = red[tid][0];
            #pragma unroll
            for (int p = 1; p < 16; ++p) m = fmaxf(m, red[tid][p]);
            mrow[tid] = m;
        }
        __syncthreads();
        // ---- p = exp(alpha - m) into f16 A-matrix buffer ----
        {
            int r = tid >> 4, p = tid & 15;
            float m = mrow[r];
            for (int j = p; j < Lsz; j += 16)
                pbuf[r][j] = (_Float16)__expf(alpha[r][j] - m);
        }
        __syncthreads();

        if (t + 1 < Tsz)  // gfx1250 global_prefetch_b8 for next-step unary
            __builtin_prefetch(&unary[((size_t)(b0 + (tid & 15))) * Tsz * Lsz +
                                      (size_t)(t + 1) * Lsz + ((tid >> 4) << 4)], 0, 1);

        // ---- GEMM: acc[16 x Lsz] = p[16 x Lsz] * expT^T ----
        // Two 16-wide column tiles per wave; A fragment shared across tiles.
        v8f acc0 = {0.f, 0.f, 0.f, 0.f, 0.f, 0.f, 0.f, 0.f};
        v8f acc1 = {0.f, 0.f, 0.f, 0.f, 0.f, 0.f, 0.f, 0.f};
        #pragma unroll
        for (int kb = 0; kb < 8; ++kb) {
            const int kbase = kb * 32;
            v16h afr, bfr0, bfr1;
            const _Float16* ap  = &pbuf[arow][kbase + koff];
            const _Float16* bp0 = &expT[(size_t)ncol0 * Lsz + kbase + bko];
            const _Float16* bp1 = &expT[(size_t)ncol1 * Lsz + kbase + bko];
            #pragma unroll
            for (int e = 0; e < 8; ++e) { afr[e] = ap[e]; afr[e + 8] = ap[16 + e]; }
            #pragma unroll
            for (int e = 0; e < 8; ++e) { bfr0[e] = bp0[e]; bfr0[e + 8] = bp0[16 + e]; bfr1[e] = bp1[e]; bfr1[e + 8] = bp1[16 + e]; }
            acc0 = __builtin_amdgcn_wmma_f32_16x16x32_f16(
                       false, afr, false, bfr0, (short)0, acc0, false, false);
            acc1 = __builtin_amdgcn_wmma_f32_16x16x32_f16(
                       false, afr, false, bfr1, (short)0, acc1, false, false);
        }
        // C/D layout: VGPR r -> M = rbase + r; N = ncolX
        #pragma unroll
        for (int r = 0; r < 8; ++r) {
            const int row = rbase + r;
            if (t < slen[row]) {
                const size_t ub = (size_t)(b0 + row) * Tsz * Lsz + (size_t)t * Lsz;
                alpha[row][ncol0] = mrow[row] + __logf(acc0[r]) + unary[ub + ncol0];
                alpha[row][ncol1] = mrow[row] + __logf(acc1[r]) + unary[ub + ncol1];
            }
        }
        __syncthreads();   // alpha writes visible before next row-max
    }

    // fwd[b] = logsumexp_i(alpha[b,i] + trans[end_idx, i])
    if (tid < 16) {
        const float* te = trans + (size_t)end_idx * Lsz;
        float m = -3.4e38f;
        for (int j = 0; j < Lsz; ++j) m = fmaxf(m, alpha[tid][j] + te[j]);
        float s = 0.f;
        for (int j = 0; j < Lsz; ++j) s += __expf(alpha[tid][j] + te[j] - m);
        fwd_out[b0 + tid] = m + __logf(s);
    }
}

// ---------------------------------------------------------------------------
// Viterbi forward (max-plus; no matrix unit for this semiring -> VALU).
// One block per batch row; full-precision f32 transitions resident in LDS
// (256 KB -- exercises the CDNA5 320 KB/WGP LDS budget); bp as uint8.
// float4 vectorized inner loop (ds_load_b128), first-max argmax semantics.
// ---------------------------------------------------------------------------
__global__ __launch_bounds__(256)
void crf_viterbi(const float* __restrict__ unary,
                 const float* __restrict__ trans,
                 const int*   __restrict__ lengths,
                 const int*   __restrict__ sidx_p,
                 const int*   __restrict__ eidx_p,
                 unsigned char* __restrict__ bp,     // [T][B][L]
                 int*   __restrict__ best_out,       // [B]
                 float* __restrict__ pscore_out)     // [B]
{
    __shared__ float tl[Lsz * Lsz];   // 256 KB, exact f32 transitions
    __shared__ float alpha[Lsz];
    __shared__ float tred[Lsz];
    __shared__ int   targ[Lsz];

    const int tid = threadIdx.x;   // == i (to-label), 0..255
    const int b   = blockIdx.x;
    const int start_idx = sidx_p[0];
    const int end_idx   = eidx_p[0];

    for (int idx = tid; idx < Lsz * Lsz; idx += 256)
        tl[idx] = trans[idx];
    alpha[tid] = (tid == start_idx) ? 0.0f : NEG_INF;
    __syncthreads();

    const int len = lengths[b];
    const v4f* tv4 = (const v4f*)&tl[(size_t)tid * Lsz];  // 1 KB-aligned row
    const v4f* av4 = (const v4f*)alpha;

    for (int t = 0; t < Tsz; ++t) {
        float bestv = -3.4e38f;
        int   bestj = 0;
        #pragma unroll 4
        for (int jb = 0; jb < Lsz / 4; ++jb) {
            v4f tv = tv4[jb];
            v4f av = av4[jb];
            #pragma unroll
            for (int e = 0; e < 4; ++e) {
                float s = av[e] + tv[e];
                if (s > bestv) { bestv = s; bestj = jb * 4 + e; }  // first-max
            }
        }
        bp[((size_t)t * Bsz + b) * Lsz + tid] = (unsigned char)bestj;
        float vit = bestv + unary[(size_t)b * Tsz * Lsz + (size_t)t * Lsz + tid];
        __syncthreads();
        if (t < len) alpha[tid] = vit;
        __syncthreads();
    }

    // terminal argmax of alpha + trans[end_idx, :]
    tred[tid] = alpha[tid] + tl[(size_t)end_idx * Lsz + tid];
    targ[tid] = tid;
    __syncthreads();
    for (int off = 128; off > 0; off >>= 1) {
        if (tid < off) {
            if (tred[tid + off] > tred[tid]) {        // strict > keeps lower index
                tred[tid] = tred[tid + off];
                targ[tid] = targ[tid + off];
            }
        }
        __syncthreads();
    }
    if (tid == 0) { best_out[b] = targ[0]; pscore_out[b] = tred[0]; }
}

// ---------------------------------------------------------------------------
// Backtrace: one thread per batch row, exact replica of reference scan.
// ---------------------------------------------------------------------------
__global__ void crf_backtrace(const unsigned char* __restrict__ bp,
                              const int* __restrict__ lengths,
                              const int* __restrict__ best,
                              float* __restrict__ paths_out)  // [B][T] as floats
{
    const int b = threadIdx.x;
    if (b >= Bsz) return;
    const int len = lengths[b];
    int cur = best[b];
    for (int t = Tsz - 1; t >= 0; --t) {
        int from_bp = (t + 1 < Tsz) ? (int)bp[((size_t)(t + 1) * Bsz + b) * Lsz + cur] : 0;
        if (t == len - 1)      cur = best[b];
        else if (t < len - 1)  cur = from_bp;
        paths_out[(size_t)b * Tsz + t] = (float)cur;
    }
}

// ---------------------------------------------------------------------------
// Gold path score: one thread per batch row.
// ---------------------------------------------------------------------------
__global__ void crf_gold(const float* __restrict__ unary,
                         const float* __restrict__ trans,
                         const int*   __restrict__ tags,
                         const int*   __restrict__ lengths,
                         const int*   __restrict__ sidx_p,
                         const int*   __restrict__ eidx_p,
                         float* __restrict__ gold_out)
{
    const int b = threadIdx.x;
    if (b >= Bsz) return;
    const int start_idx = sidx_p[0];
    const int end_idx   = eidx_p[0];
    const int len = lengths[b];
    int prev = start_idx;
    float s = 0.f;
    for (int t = 0; t < len; ++t) {
        int tg = tags[(size_t)b * Tsz + t];
        s += trans[(size_t)tg * Lsz + prev] +
             unary[(size_t)b * Tsz * Lsz + (size_t)t * Lsz + tg];
        prev = tg;
    }
    int last = tags[(size_t)b * Tsz + (len - 1)];
    s += trans[(size_t)end_idx * Lsz + last];
    gold_out[b] = s;
}

// ---------------------------------------------------------------------------
// loss = sum(fwd - gold) / sum(lengths)
// ---------------------------------------------------------------------------
__global__ void crf_loss(const float* __restrict__ fwd,
                         const float* __restrict__ gold,
                         const int*   __restrict__ lengths,
                         float* __restrict__ loss_out)
{
    __shared__ float s1[Bsz];
    __shared__ int   s2[Bsz];
    const int b = threadIdx.x;
    s1[b] = fwd[b] - gold[b];
    s2[b] = lengths[b];
    __syncthreads();
    for (int off = 64; off > 0; off >>= 1) {
        if (b < off) { s1[b] += s1[b + off]; s2[b] += s2[b + off]; }
        __syncthreads();
    }
    if (b == 0) loss_out[0] = s1[0] / (float)s2[0];
}

// ---------------------------------------------------------------------------
extern "C" void kernel_launch(void* const* d_in, const int* in_sizes, int n_in,
                              void* d_out, int out_size, void* d_ws, size_t ws_size,
                              hipStream_t stream) {
    const float* unary   = (const float*)d_in[0];   // [B,T,L]
    const float* trans   = (const float*)d_in[1];   // [L,L]
    const int*   tags    = (const int*)  d_in[2];   // [B,T]
    const int*   lengths = (const int*)  d_in[3];   // [B]
    const int*   sidx    = (const int*)  d_in[4];   // scalar
    const int*   eidx    = (const int*)  d_in[5];   // scalar

    float* out = (float*)d_out;
    float* out_loss   = out;                 // [1]
    float* out_paths  = out + 1;             // [B*T]
    float* out_pscore = out + 1 + Bsz * Tsz; // [B]

    // Workspace layout
    float* ws_f      = (float*)d_ws;
    float* ws_fwd    = ws_f;            // 128
    float* ws_gold   = ws_f + 128;      // 128
    int*   ws_best   = (int*)(ws_f + 256); // 128
    unsigned char* ws_bp = (unsigned char*)d_ws + 2048;  // T*B*L bytes = 16 MiB

    crf_forward_wmma<<<Bsz / 16, 256, 0, stream>>>(unary, trans, lengths, sidx, eidx, ws_fwd);
    crf_gold<<<1, Bsz, 0, stream>>>(unary, trans, tags, lengths, sidx, eidx, ws_gold);
    crf_viterbi<<<Bsz, 256, 0, stream>>>(unary, trans, lengths, sidx, eidx,
                                         ws_bp, ws_best, out_pscore);
    crf_backtrace<<<1, Bsz, 0, stream>>>(ws_bp, lengths, ws_best, out_paths);
    crf_loss<<<1, Bsz, 0, stream>>>(ws_fwd, ws_gold, lengths, out_loss);
}
